// GCNStreamModule_38104949850543
// MI455X (gfx1250) — matrix operand order repaired and checked by reading, not compiled, not run on hardware
//
#include <hip/hip_runtime.h>

typedef __attribute__((ext_vector_type(16))) __bf16 bf16x16;
typedef __attribute__((ext_vector_type(8)))  __bf16 bf16x8;
typedef __attribute__((ext_vector_type(2)))  __bf16 bf16x2;
typedef __attribute__((ext_vector_type(8)))  float  f32x8;

#define BM 128
#define BN 64
#define BK 32
#define LDP (BK + 8)   // LDS row pitch in bf16 elems (80 B, 16B-aligned rows)

__device__ __forceinline__ bf16x8 pack8(float4 a, float4 b) {
    bf16x8 r;
    r[0] = (__bf16)a.x; r[1] = (__bf16)a.y; r[2] = (__bf16)a.z; r[3] = (__bf16)a.w;
    r[4] = (__bf16)b.x; r[5] = (__bf16)b.y; r[6] = (__bf16)b.z; r[7] = (__bf16)b.w;
    return r;
}

__device__ __forceinline__ bf16x16 cat16(bf16x8 lo, bf16x8 hi) {
    return __builtin_shufflevector(lo, hi, 0, 1, 2, 3, 4, 5, 6, 7,
                                           8, 9, 10, 11, 12, 13, 14, 15);
}

// ---------------------------------------------------------------------------
// C = A[MxK] @ B[KxN] + bias. fp32 in/out, bf16 WMMA f32-accumulate.
// 256 threads = 8 waves; block tile 128x64; wave tile 16x64 (4 wmma/chunk).
// Ping-pong LDS (one barrier per chunk), register prefetch of chunk k+1
// overlapping the chunk-k WMMAs, pointer-increment addressing (clamps are
// k-invariant). OOB rows/cols: address clamp only — garbage D rows/cols are
// never stored and WMMA raises no exceptions.
// Requires K % 64 == 0 (K = 1024 / 256 here). bias must be a valid pointer.
// ---------------------------------------------------------------------------
__global__ __launch_bounds__(256) void gemm_bf16_nn(
    const float* __restrict__ A, const float* __restrict__ B,
    const float* __restrict__ bias, float* __restrict__ C,
    int M, int N, int K)
{
    __shared__ __bf16 As0[BM][LDP], As1[BM][LDP];
    __shared__ __bf16 Bs0[BN][LDP], Bs1[BN][LDP];   // transposed: Bs[n][k]

    const int tid  = threadIdx.x;
    const int lane = tid & 31;
    const int wave = tid >> 5;       // 0..7 -> 16-row strip

    const int row0 = blockIdx.y * BM;
    const int col0 = blockIdx.x * BN;

    f32x8 acc[4] = {{}, {}, {}, {}};

    const int m_loc = wave * 16 + (lane & 15);
    const int kbA   = (lane >> 4) * 8;    // A frag K sub-base {0,8}
    const int koB   = (lane >> 4) * 8;    // B frag K sub-base {0,8}

    // ---- per-task base pointers (clamps are k-invariant) ----
    const float* pa[2];
    int ari[2], aci[2];
    #pragma unroll
    for (int i = 0; i < 2; ++i) {
        const int T  = tid + 256 * i;
        ari[i] = T >> 2;                     // 0..127
        aci[i] = (T & 3) * 8;                // 0,8,16,24
        int gr = row0 + ari[i];
        gr = gr < M ? gr : (M - 1);
        pa[i] = A + (size_t)gr * K + aci[i];
    }
    const float* pb[4];
    int bni[4], bki[4];
    #pragma unroll
    for (int j = 0; j < 4; ++j) {
        const int T = tid + 256 * j;
        bni[j] = T & 63;                     // lane-consecutive -> coalesced
        bki[j] = (T >> 6) * 2;               // even k within chunk
        int gn = col0 + bni[j];
        gn = gn < N ? gn : (N - 1);
        pb[j] = B + (size_t)bki[j] * N + gn;
    }

    float4 ra[2][2];   // A prefetch regs
    float  rb[4][2];   // B prefetch regs (k-pair)

    auto fetch = [&]() {   // loads current chunk, advances pointers by BK
        #pragma unroll
        for (int i = 0; i < 2; ++i) {
            const float4* s4 = (const float4*)pa[i];
            ra[i][0] = s4[0];
            ra[i][1] = s4[1];
            pa[i] += BK;
        }
        #pragma unroll
        for (int j = 0; j < 4; ++j) {
            rb[j][0] = pb[j][0];
            rb[j][1] = pb[j][N];
            pb[j] += (size_t)BK * N;
        }
    };

    auto stage = [&](__bf16 (&Asb)[BM][LDP], __bf16 (&Bsb)[BN][LDP]) {
        #pragma unroll
        for (int i = 0; i < 2; ++i)
            *(bf16x8*)&Asb[ari[i]][aci[i]] = pack8(ra[i][0], ra[i][1]);
        #pragma unroll
        for (int j = 0; j < 4; ++j) {
            bf16x2 pr; pr[0] = (__bf16)rb[j][0]; pr[1] = (__bf16)rb[j][1];
            *(bf16x2*)&Bsb[bni[j]][bki[j]] = pr;
        }
    };

    auto compute = [&](const __bf16 (&Asb)[BM][LDP], const __bf16 (&Bsb)[BN][LDP]) {
        const bf16x8* ap = (const bf16x8*)&Asb[m_loc][kbA];
        const bf16x16 afrag = cat16(ap[0], ap[2]);   // K {kbA..+7, kbA+16..+23}
        #pragma unroll
        for (int nn = 0; nn < 4; ++nn) {
            const bf16x8* bp = (const bf16x8*)&Bsb[nn * 16 + (lane & 15)][koB];
            const bf16x16 bfrag = cat16(bp[0], bp[2]); // K {koB..+7, koB+16..+23}
            acc[nn] = __builtin_amdgcn_wmma_f32_16x16x32_bf16(
                false, afrag, false, bfrag, (short)0, acc[nn], false, false);
        }
    };

    // ---- software pipeline: 1 barrier per chunk, static ping-pong ----
    fetch();              // chunk 0
    stage(As0, Bs0);
    __syncthreads();
    for (int k0 = 0; k0 < K; k0 += 2 * BK) {
        fetch();                          // chunk k0+32 (always exists: K%64==0)
        compute(As0, Bs0);
        stage(As1, Bs1);
        __syncthreads();

        const bool more = (k0 + 2 * BK) < K;
        if (more) fetch();                // chunk k0+64
        compute(As1, Bs1);
        if (more) stage(As0, Bs0);
        __syncthreads();
    }

    // ---- epilogue: D layout lane(&15)=N, vgpr v -> M = v + 8*(lane>>4) ----
    #pragma unroll
    for (int nn = 0; nn < 4; ++nn) {
        const int gn = col0 + nn * 16 + (lane & 15);
        if (gn < N) {
            const float badd = bias[gn];
            #pragma unroll
            for (int v = 0; v < 8; ++v) {
                const int gm = row0 + wave * 16 + v + 8 * (lane >> 4);
                if (gm < M) C[(size_t)gm * N + gn] = acc[nn][v] + badd;
            }
        }
    }
}

// ---------------------------------------------------------------------------
// Out[M,N] = sigmoid( Xq[M,K] @ F[N,K]^T ).  K=256.  B rows are contiguous
// f2 rows (L2-resident) -> direct float4 global loads, no LDS. Non-temporal
// output stores keep the 410 MB result stream from evicting f2 from L2.
// Each wave owns a 16x64 tile (4 accumulators). OOB cols: address clamp only.
// ---------------------------------------------------------------------------
__global__ __launch_bounds__(256) void gemm_nt_sigmoid(
    const float* __restrict__ Xq, const float* __restrict__ F,
    float* __restrict__ Out, int M, int N, int K, int tilesN)
{
    const int lane = threadIdx.x & 31;
    const int wave = threadIdx.x >> 5;
    const long tile = (long)blockIdx.x * 8 + wave;
    const long totalTiles = (long)(M >> 4) * tilesN;
    if (tile >= totalTiles) return;          // wave-uniform exit
    const int tn = (int)(tile % tilesN);
    const int tm = (int)(tile / tilesN);

    f32x8 acc[4] = {{}, {}, {}, {}};

    const int mrow = tm * 16 + (lane & 15);       // always < M (M=2048)
    const int kbA  = (lane >> 4) * 8;
    const int koB  = (lane >> 4) * 8;

    for (int k0 = 0; k0 < K; k0 += 32) {
        // A fragment: two pairs of float4 (K {kbA..+7, kbA+16..+23})
        const float4* a4 = (const float4*)(Xq + (size_t)mrow * K + k0 + kbA);
        const bf16x16 afrag = cat16(pack8(a4[0], a4[1]), pack8(a4[4], a4[5]));

        #pragma unroll
        for (int nn = 0; nn < 4; ++nn) {
            int n = tn * 64 + nn * 16 + (lane & 15);
            n = n < N ? n : (N - 1);              // address clamp only
            const float4* b4 = (const float4*)(F + (size_t)n * K + k0 + koB);
            const bf16x16 bfrag = cat16(pack8(b4[0], b4[1]), pack8(b4[4], b4[5]));
            acc[nn] = __builtin_amdgcn_wmma_f32_16x16x32_bf16(
                false, afrag, false, bfrag, (short)0, acc[nn], false, false);
        }
    }

    #pragma unroll
    for (int nn = 0; nn < 4; ++nn) {
        const int gn = tn * 64 + nn * 16 + (lane & 15);
        if (gn < N) {
            #pragma unroll
            for (int v = 0; v < 8; ++v) {
                const int gm = tm * 16 + v + 8 * (lane >> 4);
                const float s = 1.0f / (1.0f + __expf(-acc[nn][v]));
                __builtin_nontemporal_store(s, &Out[(size_t)gm * N + gn]);
            }
        }
    }
}

// ---------------------------------------------------------------------------
// spmm scatter: out[row[e],:] += val[e] * dense[col[e],:]  (256 feats)
// One wave per edge; native fp32 global atomics (rows are L2-resident).
// ---------------------------------------------------------------------------
__global__ __launch_bounds__(256) void spmm_scatter(
    const int* __restrict__ erow, const int* __restrict__ ecol,
    const float* __restrict__ eval, const float* __restrict__ dense,
    float* __restrict__ out, int nE)
{
    const int e = blockIdx.x * 8 + (threadIdx.x >> 5);
    if (e >= nE) return;
    const int lane = threadIdx.x & 31;
    const int r = erow[e];
    const int c = ecol[e];
    const float v = eval[e];
    const float* src = dense + (size_t)c * 256;
    float* dst = out + (size_t)r * 256;
    #pragma unroll
    for (int i = 0; i < 8; ++i) {
        const int f = lane + i * 32;
        __hip_atomic_fetch_add(&dst[f], v * src[f],
                               __ATOMIC_RELAXED, __HIP_MEMORY_SCOPE_AGENT);
    }
}

__global__ __launch_bounds__(256) void fill_zero4(float4* __restrict__ p, long n4)
{
    const long i = (long)blockIdx.x * blockDim.x + threadIdx.x;
    if (i < n4) p[i] = make_float4(0.f, 0.f, 0.f, 0.f);
}

__global__ __launch_bounds__(256) void bias_act(
    float* __restrict__ buf, const float* __restrict__ bias, long n, int relu)
{
    const long i = (long)blockIdx.x * blockDim.x + threadIdx.x;
    if (i >= n) return;
    float v = buf[i] + bias[(int)(i & 255)];
    if (relu) v = fmaxf(v, 0.0f);
    buf[i] = v;
}

// ---------------------------------------------------------------------------
extern "C" void kernel_launch(void* const* d_in, const int* in_sizes, int n_in,
                              void* d_out, int out_size, void* d_ws, size_t ws_size,
                              hipStream_t stream)
{
    const float* x      = (const float*)d_in[0];   // [2048,256]
    const float* X      = (const float*)d_in[1];   // [50000,1024]
    const float* W_comp = (const float*)d_in[2];   // [1024,256]
    const float* b_comp = (const float*)d_in[3];   // [256]
    const float* W1     = (const float*)d_in[4];   // [256,256]
    const float* b1     = (const float*)d_in[5];   // [256]
    const float* W2     = (const float*)d_in[6];   // [256,256]
    const float* b2     = (const float*)d_in[7];   // [256]
    const float* eval   = (const float*)d_in[8];   // [800000]
    const int*   erow   = (const int*)d_in[9];     // [800000]
    const int*   ecol   = (const int*)d_in[10];    // [800000]
    float* out = (float*)d_out;                    // [2048, 50000]

    const int Nn = 50000, E = 800000, FEAT = 1024, OC = 256, Bq = 2048;
    const long bufElems = (long)Nn * OC;           // 12.8M floats

    float* bufA = (float*)d_ws;                    // 51.2 MB
    float* bufB = bufA + bufElems;                 // 51.2 MB
    float* bufZ = bufB + bufElems;                 // 1 KB zero bias

    const dim3 blk(256);
    const dim3 gridGemm((OC + BN - 1) / BN, (Nn + BM - 1) / BM);
    const int fillBlocks = (int)((bufElems / 4 + 255) / 256);
    const int eaBlocks   = (int)((bufElems + 255) / 256);
    const int spBlocks   = (E + 7) / 8;

    // zero-bias stripe (256 floats)
    fill_zero4<<<1, 64, 0, stream>>>((float4*)bufZ, 64);

    // 1) f0 = X @ W_comp + b_comp            -> bufA
    gemm_bf16_nn<<<gridGemm, blk, 0, stream>>>(X, W_comp, b_comp, bufA, Nn, OC, FEAT);

    // 2) g1 = f0 @ W1                        -> bufB
    gemm_bf16_nn<<<gridGemm, blk, 0, stream>>>(bufA, W1, bufZ, bufB, Nn, OC, OC);

    // 3) f1 = relu(spmm(g1) + b1)            -> bufA
    fill_zero4<<<fillBlocks, blk, 0, stream>>>((float4*)bufA, bufElems / 4);
    spmm_scatter<<<spBlocks, blk, 0, stream>>>(erow, ecol, eval, bufB, bufA, E);
    bias_act<<<eaBlocks, blk, 0, stream>>>(bufA, b1, bufElems, 1);

    // 4) g2 = f1 @ W2                        -> bufB
    gemm_bf16_nn<<<gridGemm, blk, 0, stream>>>(bufA, W2, bufZ, bufB, Nn, OC, OC);

    // 5) f2 = spmm(g2) + b2                  -> bufA
    fill_zero4<<<fillBlocks, blk, 0, stream>>>((float4*)bufA, bufElems / 4);
    spmm_scatter<<<spBlocks, blk, 0, stream>>>(erow, ecol, eval, bufB, bufA, E);
    bias_act<<<eaBlocks, blk, 0, stream>>>(bufA, b2, bufElems, 0);

    // 6) out = sigmoid(x @ f2^T)
    const int tilesN = (Nn + 63) / 64;                   // 782
    const long totalTiles = (long)(Bq / 16) * tilesN;    // 100096
    const int ntBlocks = (int)((totalTiles + 7) / 8);
    gemm_nt_sigmoid<<<ntBlocks, blk, 0, stream>>>(x, bufA, out, Bq, Nn, OC, tilesN);
}
